// VAST_87316685128005
// MI455X (gfx1250) — matrix operand order repaired and checked by reading, not compiled, not run on hardware
//
#include <hip/hip_runtime.h>
#include <hip/hip_bf16.h>
#include <math.h>

// ---------- CDNA5 WMMA plumbing (gfx1250, wave32) ----------
typedef __attribute__((ext_vector_type(16))) __bf16 bf16x16;
typedef __attribute__((ext_vector_type(8)))  __bf16 bf16x8;
typedef __attribute__((ext_vector_type(8)))  float  f32x8;

__device__ __forceinline__ __bf16 f2bf(float x) {
    unsigned u = __builtin_bit_cast(unsigned, x);
    unsigned r = (u + 0x7FFFu + ((u >> 16) & 1u)) >> 16;
    unsigned short s = (unsigned short)r;
    return __builtin_bit_cast(__bf16, s);
}
__device__ __forceinline__ float bf2f(__bf16 x) {
    unsigned short s = __builtin_bit_cast(unsigned short, x);
    unsigned u = ((unsigned)s) << 16;
    return __builtin_bit_cast(float, u);
}

// A fragment (16x32, row per lane): elems 0..7 = [row][kb..kb+7], 8..15 = [row][kb+16..kb+23]
__device__ __forceinline__ bf16x16 frag_ld_a(const __bf16* p) {
    bf16x8 lo = *(const bf16x8*)(p);
    bf16x8 hi = *(const bf16x8*)(p + 16);
    bf16x16 r;
#pragma unroll
    for (int i = 0; i < 8; ++i) { r[i] = lo[i]; r[i + 8] = hi[i]; }
    return r;
}
// B fragment (32x16, col per lane): elems 0..7 = [col][kh*8..kh*8+7], 8..15 = [col][kh*8+16..kh*8+23]
__device__ __forceinline__ bf16x16 frag_ld_b(const __bf16* p) {
    p -= ((threadIdx.x & 31) >> 4) << 3;
    bf16x8 lo = *(const bf16x8*)(p);
    bf16x8 hi = *(const bf16x8*)(p + 16);
    bf16x16 r;
#pragma unroll
    for (int i = 0; i < 8; ++i) { r[i] = lo[i]; r[i + 8] = hi[i]; }
    return r;
}
__device__ __forceinline__ f32x8 wmma_bf16(bf16x16 a, bf16x16 b, f32x8 c) {
    return __builtin_amdgcn_wmma_f32_16x16x32_bf16(false, a, false, b, (short)0, c, false, false);
}

// ---------- gfx1250 async global->LDS copy (ASYNCcnt-tracked DMA) ----------
__device__ __forceinline__ unsigned lds_off_u32(const void* p) {
    return (unsigned)(size_t)p;   // low 32 bits of a generic LDS pointer = LDS offset
}
__device__ __forceinline__ void async_b128(unsigned lds_off, const void* gaddr) {
    asm volatile("global_load_async_to_lds_b128 %0, %1, off"
                 :: "v"(lds_off), "v"(gaddr) : "memory");
}
__device__ __forceinline__ void wait_async0() {
    asm volatile("s_wait_asynccnt 0x0" ::: "memory");
}
__device__ __forceinline__ void wait_async3() {
    asm volatile("s_wait_asynccnt 0x3" ::: "memory");
}

// ---------- f32 -> bf16 convert (weights/states, once per launch) ----------
__global__ void __launch_bounds__(256)
cvt_bf16_kernel(const float* __restrict__ src, __bf16* __restrict__ dst, int n) {
    int i = (blockIdx.x * 256 + threadIdx.x) * 4;
    if (i + 3 < n) {
        float4 v = *(const float4*)(src + i);
        dst[i + 0] = f2bf(v.x); dst[i + 1] = f2bf(v.y);
        dst[i + 2] = f2bf(v.z); dst[i + 3] = f2bf(v.w);
    }
}

// ---------- GEMM: C[M,N] = (Abf[M,K] x Wbf[N,K]^T + bias[N]) * alpha ----------
// 128x64 block tile, 8 waves of 32x32 (4 WMMA / K-step), double-buffered LDS,
// tiles streamed with GLOBAL_LOAD_ASYNC_TO_LDS_B128.
#define GLDK 40   // bf16 row stride: 80B (16B aligned), spreads LDS banks

template <bool GELU, bool OUT_BF16>
__global__ void __launch_bounds__(256)
gemm_bf16_kernel(const __bf16* __restrict__ A, const __bf16* __restrict__ W,
                 const float* __restrict__ bias, void* __restrict__ Cout,
                 int M, int N, int K, float alpha) {
    __shared__ __bf16 AsBuf[2][128 * GLDK];
    __shared__ __bf16 WsBuf[2][64 * GLDK];

    const int tid  = threadIdx.x;
    const int lane = tid & 31;
    const int wave = tid >> 5;
    const int wm   = wave & 3;            // 4 row groups of 32
    const int wn   = wave >> 2;           // 2 col groups of 32
    const int m0   = blockIdx.y * 128;
    const int n0   = blockIdx.x * 64;
    const int fr   = lane & 15;
    const int kb   = (lane >> 4) << 3;    // A-frag K base
    const int khb  = (lane >> 4) << 4;    // B-frag K base

    const unsigned asOff[2] = { lds_off_u32(&AsBuf[0][0]), lds_off_u32(&AsBuf[1][0]) };
    const unsigned wsOff[2] = { lds_off_u32(&WsBuf[0][0]), lds_off_u32(&WsBuf[1][0]) };

    // per-thread copy chunks: A tile 128x32 = 512 x 16B (2/thread), W tile 64x32 = 256 x 16B (1/thread)
    const int ar0 = tid >> 2,          ac0 = (tid & 3) << 3;
    const int ar1 = (tid + 256) >> 2,  ac1 = ac0;
    const int wr  = tid >> 2,          wc  = ac0;

    auto copy_tile = [&](int buf, int k0) {
        async_b128(asOff[buf] + (unsigned)(ar0 * GLDK + ac0) * 2,
                   A + (size_t)(m0 + ar0) * K + k0 + ac0);
        async_b128(asOff[buf] + (unsigned)(ar1 * GLDK + ac1) * 2,
                   A + (size_t)(m0 + ar1) * K + k0 + ac1);
        async_b128(wsOff[buf] + (unsigned)(wr * GLDK + wc) * 2,
                   W + (size_t)(n0 + wr) * K + k0 + wc);
    };

    f32x8 acc00 = {}, acc01 = {}, acc10 = {}, acc11 = {};

    const int nsteps = K >> 5;
    copy_tile(0, 0);
    for (int s = 0; s < nsteps; ++s) {
        const int cur = s & 1;
        __syncthreads();                          // everyone done reading buf[cur^1]
        if (s + 1 < nsteps) { copy_tile(cur ^ 1, (s + 1) << 5); wait_async3(); }
        else                { wait_async0(); }
        __syncthreads();                          // LDS tile visible to all waves

        const __bf16* As = &AsBuf[cur][0];
        const __bf16* Ws = &WsBuf[cur][0];
        bf16x16 a0 = frag_ld_a(As + (wm * 32 + fr) * GLDK + kb);
        bf16x16 a1 = frag_ld_a(As + (wm * 32 + 16 + fr) * GLDK + kb);
        bf16x16 b0 = frag_ld_b(Ws + (wn * 32 + fr) * GLDK + khb);
        bf16x16 b1 = frag_ld_b(Ws + (wn * 32 + 16 + fr) * GLDK + khb);
        acc00 = wmma_bf16(a0, b0, acc00);
        acc01 = wmma_bf16(a0, b1, acc01);
        acc10 = wmma_bf16(a1, b0, acc10);
        acc11 = wmma_bf16(a1, b1, acc11);
    }

    const int mbase = m0 + wm * 32 + ((lane >> 4) << 3);
    const int nbase = n0 + wn * 32 + (lane & 15);
    const float bA = bias[nbase], bB = bias[nbase + 16];
    float* Cf = (float*)Cout;
    __bf16* Cb = (__bf16*)Cout;
#pragma unroll
    for (int r = 0; r < 8; ++r) {
        float v00 = (acc00[r] + bA) * alpha;
        float v01 = (acc01[r] + bB) * alpha;
        float v10 = (acc10[r] + bA) * alpha;
        float v11 = (acc11[r] + bB) * alpha;
        if (GELU) {
            v00 = 0.5f * v00 * (1.0f + erff(v00 * 0.70710678f));
            v01 = 0.5f * v01 * (1.0f + erff(v01 * 0.70710678f));
            v10 = 0.5f * v10 * (1.0f + erff(v10 * 0.70710678f));
            v11 = 0.5f * v11 * (1.0f + erff(v11 * 0.70710678f));
        }
        size_t r0 = (size_t)(mbase + r) * N;
        size_t r1 = (size_t)(mbase + 16 + r) * N;
        if (OUT_BF16) {
            Cb[r0 + nbase] = f2bf(v00); Cb[r0 + nbase + 16] = f2bf(v01);
            Cb[r1 + nbase] = f2bf(v10); Cb[r1 + nbase + 16] = f2bf(v11);
        } else {
            Cf[r0 + nbase] = v00; Cf[r0 + nbase + 16] = v01;
            Cf[r1 + nbase] = v10; Cf[r1 + nbase + 16] = v11;
        }
    }
}

// ---------- Flash attention with gated positional bias ----------
// q stored bf16 pre-scaled (alpha=0.125 folds the x32 logit scale); probs =
// softmax(q32.k^T + gA1[t]*bias); Q/K tiles async-DMA'd into LDS, V transposed.
#define ALD 72

__global__ void __launch_bounds__(256)
attn_kernel(const __bf16* __restrict__ q, const __bf16* __restrict__ kbuf,
            const __bf16* __restrict__ vbuf, const float* __restrict__ pb,
            const float* __restrict__ gw, const float* __restrict__ gb,
            const float* __restrict__ ga, __bf16* __restrict__ ctx) {
    __shared__ __bf16 Qs[64][ALD];
    __shared__ __bf16 Ks[64][ALD];
    __shared__ __bf16 Vt[64][ALD];   // [d][s]
    __shared__ __bf16 Ps[64][ALD];
    __shared__ float  glog[64][8];
    __shared__ float  Sb[64][68];
    __shared__ float  gA1[64], rmax[64], rsum[64], rscale[64];
    __shared__ float  red[256];

    const int tid = threadIdx.x, lane = tid & 31, wave = tid >> 5;
    const int wm = wave & 3, wn = wave >> 2;
    const int fr = lane & 15;
    const int kb = (lane >> 4) << 3;
    const int khb = (lane >> 4) << 4;
    const int n  = blockIdx.y;
    const int bb = n / 12, hh = n % 12;
    const int tq0 = blockIdx.x * 64;

    const unsigned qsOff = lds_off_u32(&Qs[0][0]);
    const unsigned ksOff = lds_off_u32(&Ks[0][0]);
    // 64x64 bf16 tile = 512 x 16B chunks, 2 per thread
    const int cr0 = tid >> 3,          cc0 = (tid & 7) << 3;
    const int cr1 = (tid + 256) >> 3,  cc1 = cc0;

    // ---- kick off Q tile DMA (q already carries the x32 logit scale)
    async_b128(qsOff + (unsigned)(cr0 * ALD + cc0) * 2,
               q + ((size_t)(tq0 + cr0) * 4 + bb) * 768 + hh * 64 + cc0);
    async_b128(qsOff + (unsigned)(cr1 * ALD + cc1) * 2,
               q + ((size_t)(tq0 + cr1) * 4 + bb) * 768 + hh * 64 + cc1);

    // ---- gate: glog = (q_stored*8).gw + gb ; gA1 = sig(a)*(sig(b)*grep_a-1)+2
#pragma unroll
    for (int it = 0; it < 2; ++it) {
        int idx = tid + it * 256;
        int row = idx >> 3, j = idx & 7;
        const __bf16* qrow = q + ((size_t)(tq0 + row) * 4 + bb) * 768 + hh * 64;
        float dot = 0.f;
        for (int d = 0; d < 64; ++d) dot += bf2f(qrow[d]) * gw[j * 64 + d];
        glog[row][j] = dot * 8.0f + gb[j];
    }
    __syncthreads();
    if (tid < 64) {
        float sa  = glog[tid][0] + glog[tid][1] + glog[tid][2] + glog[tid][3];
        float sbv = glog[tid][4] + glog[tid][5] + glog[tid][6] + glog[tid][7];
        float gaV = 1.f / (1.f + __expf(-sa));
        float gbV = 1.f / (1.f + __expf(-sbv));
        gA1[tid] = gaV * (gbV * ga[hh] - 1.0f) + 2.0f;
        rmax[tid] = -1e30f; rsum[tid] = 0.f;
    }

    f32x8 o0 = {}, o1 = {};
    const int srow = tid >> 2, part = tid & 3;

    for (int s0 = 0; s0 < 1024; s0 += 64) {
        __syncthreads();
        // K tile: async DMA; V tile: load bf16 + transpose into Vt[d][s]
        async_b128(ksOff + (unsigned)(cr0 * ALD + cc0) * 2,
                   kbuf + ((size_t)(s0 + cr0) * 4 + bb) * 768 + hh * 64 + cc0);
        async_b128(ksOff + (unsigned)(cr1 * ALD + cc1) * 2,
                   kbuf + ((size_t)(s0 + cr1) * 4 + bb) * 768 + hh * 64 + cc1);
#pragma unroll
        for (int it = 0; it < 2; ++it) {
            int idx = tid + it * 256;
            int row = idx >> 3, c8 = (idx & 7) << 3;
            bf16x8 vv = *(const bf16x8*)(vbuf + ((size_t)(s0 + row) * 4 + bb) * 768 + hh * 64 + c8);
#pragma unroll
            for (int i = 0; i < 8; ++i) Vt[c8 + i][row] = vv[i];
        }
        wait_async0();
        __syncthreads();
        // ---- S = Qs . Ks^T (each wave 16x32 of 64x64)
        f32x8 sa0 = {}, sa1 = {};
#pragma unroll
        for (int kk = 0; kk < 64; kk += 32) {
            bf16x16 af = frag_ld_a(&Qs[wm * 16 + fr][kk + kb]);
            bf16x16 b0 = frag_ld_b(&Ks[wn * 32 + fr][kk + khb]);
            bf16x16 b1 = frag_ld_b(&Ks[wn * 32 + 16 + fr][kk + khb]);
            sa0 = wmma_bf16(af, b0, sa0);
            sa1 = wmma_bf16(af, b1, sa1);
        }
        {
            int mr = wm * 16 + ((lane >> 4) << 3);
            int nc = wn * 32 + (lane & 15);
#pragma unroll
            for (int r = 0; r < 8; ++r) {
                Sb[mr + r][nc]      = sa0[r];
                Sb[mr + r][nc + 16] = sa1[r];
            }
        }
        __syncthreads();
        // ---- gated bias + online softmax (4 threads/row x 16 cols)
        const float g1 = gA1[srow];
        const float* bptr = pb + ((size_t)n * 1024 + (tq0 + srow)) * 1024 + s0 + part * 16;
        if (s0 + 64 < 1024) __builtin_prefetch(bptr + 64, 0, 0);
        float lv[16];
        float lmax = -1e30f;
#pragma unroll
        for (int i = 0; i < 16; ++i) {
            float val = Sb[srow][part * 16 + i] + g1 * bptr[i];
            lv[i] = val;
            lmax = fmaxf(lmax, val);
        }
        red[tid] = lmax;
        __syncthreads();
        if (part == 0) {
            float mn = fmaxf(fmaxf(red[tid], red[tid + 1]), fmaxf(red[tid + 2], red[tid + 3]));
            mn = fmaxf(mn, rmax[srow]);
            rscale[srow] = __expf(rmax[srow] - mn);
            rmax[srow]   = mn;
            rsum[srow]  *= rscale[srow];
        }
        __syncthreads();
        const float mrow_ = rmax[srow];
        float psum = 0.f;
#pragma unroll
        for (int i = 0; i < 16; ++i) {
            float e = __expf(lv[i] - mrow_);
            psum += e;
            Ps[srow][part * 16 + i] = f2bf(e);
        }
        red[tid] = psum;
        __syncthreads();
        if (part == 0)
            rsum[srow] += red[tid] + red[tid + 1] + red[tid + 2] + red[tid + 3];
        {
            int mr = wm * 16 + ((lane >> 4) << 3);
#pragma unroll
            for (int r = 0; r < 8; ++r) {
                float f = rscale[mr + r];
                o0[r] *= f; o1[r] *= f;
            }
        }
        __syncthreads();
        // ---- O += P . V
#pragma unroll
        for (int kk = 0; kk < 64; kk += 32) {
            bf16x16 af = frag_ld_a(&Ps[wm * 16 + fr][kk + kb]);
            bf16x16 b0 = frag_ld_b(&Vt[wn * 32 + fr][kk + khb]);
            bf16x16 b1 = frag_ld_b(&Vt[wn * 32 + 16 + fr][kk + khb]);
            o0 = wmma_bf16(af, b0, o0);
            o1 = wmma_bf16(af, b1, o1);
        }
    }
    __syncthreads();
    // ---- normalize + write bf16 ctx in (T,B,D) layout (feeds Wo GEMM)
    {
        int mr = wm * 16 + ((lane >> 4) << 3);
        int nc = wn * 32 + (lane & 15);
#pragma unroll
        for (int r = 0; r < 8; ++r) {
            float inv = 1.0f / rsum[mr + r];
            size_t base = ((size_t)(tq0 + mr + r) * 4 + bb) * 768 + hh * 64;
            ctx[base + nc]      = f2bf(o0[r] * inv);
            ctx[base + nc + 16] = f2bf(o1[r] * inv);
        }
    }
}

// ---------- out = LayerNorm(xa*ascale + xb)*g + b ; optional bf16 copy ----------
template <bool WB>
__global__ void __launch_bounds__(256)
add_ln_kernel(const float* __restrict__ xa, float ascale, const float* __restrict__ xb,
              const float* __restrict__ g, const float* __restrict__ b,
              float* __restrict__ out, __bf16* __restrict__ outb) {
    const int row = blockIdx.x, tid = threadIdx.x;
    __shared__ float red[256], red2[256];
    const float* pa  = xa + (size_t)row * 768;
    const float* pbv = xb + (size_t)row * 768;
    float vals[3];
    float s = 0.f, s2 = 0.f;
#pragma unroll
    for (int i = 0; i < 3; ++i) {
        int c = tid + i * 256;
        float vv = pa[c] * ascale + pbv[c];
        vals[i] = vv; s += vv; s2 += vv * vv;
    }
    red[tid] = s; red2[tid] = s2;
    __syncthreads();
    for (int off = 128; off > 0; off >>= 1) {
        if (tid < off) { red[tid] += red[tid + off]; red2[tid] += red2[tid + off]; }
        __syncthreads();
    }
    const float mu  = red[0] * (1.0f / 768.0f);
    const float var = red2[0] * (1.0f / 768.0f) - mu * mu;
    const float inv = rsqrtf(var + 1e-5f);
#pragma unroll
    for (int i = 0; i < 3; ++i) {
        int c = tid + i * 256;
        float o = (vals[i] - mu) * inv * g[c] + b[c];
        out[(size_t)row * 768 + c] = o;
        if (WB) outb[(size_t)row * 768 + c] = f2bf(o);
    }
}

// ---------- orchestration ----------
extern "C" void kernel_launch(void* const* d_in, const int* in_sizes, int n_in,
                              void* d_out, int out_size, void* d_ws, size_t ws_size,
                              hipStream_t stream) {
    (void)in_sizes; (void)n_in; (void)out_size; (void)ws_size;
    const float* states = (const float*)d_in[0];
    const float* pbias  = (const float*)d_in[1];
    const float* Wq = (const float*)d_in[2];  const float* bq = (const float*)d_in[3];
    const float* Wk = (const float*)d_in[4];  const float* bk = (const float*)d_in[5];
    const float* Wv = (const float*)d_in[6];  const float* bv = (const float*)d_in[7];
    const float* Wo = (const float*)d_in[8];  const float* bo = (const float*)d_in[9];
    const float* gw = (const float*)d_in[10]; const float* gb = (const float*)d_in[11];
    const float* ga = (const float*)d_in[12];
    const float* l1g = (const float*)d_in[13]; const float* l1b = (const float*)d_in[14];
    const float* W1 = (const float*)d_in[15]; const float* b1 = (const float*)d_in[16];
    const float* W2 = (const float*)d_in[17]; const float* b2 = (const float*)d_in[18];
    const float* l2g = (const float*)d_in[19]; const float* l2b = (const float*)d_in[20];

    const size_t MD = (size_t)4096 * 768;       // 3,145,728
    const size_t WD = (size_t)768 * 768;        //   589,824
    const size_t WF = (size_t)768 * 3072;       // 2,359,296
    const size_t MF = (size_t)4096 * 3072;      // 12,582,912

    char* w = (char*)d_ws;
    auto alloc = [&](size_t bytes) { void* p = (void*)w; w += (bytes + 255) & ~(size_t)255; return p; };
    __bf16* statesb = (__bf16*)alloc(MD * 2);
    __bf16* Wqb = (__bf16*)alloc(WD * 2);
    __bf16* Wkb = (__bf16*)alloc(WD * 2);
    __bf16* Wvb = (__bf16*)alloc(WD * 2);
    __bf16* Wob = (__bf16*)alloc(WD * 2);
    __bf16* W1b = (__bf16*)alloc(WF * 2);
    __bf16* W2b = (__bf16*)alloc(WF * 2);
    __bf16* qb  = (__bf16*)alloc(MD * 2);
    __bf16* kb  = (__bf16*)alloc(MD * 2);
    __bf16* vb  = (__bf16*)alloc(MD * 2);
    __bf16* ctxb= (__bf16*)alloc(MD * 2);
    __bf16* x1b = (__bf16*)alloc(MD * 2);
    __bf16* h1b = (__bf16*)alloc(MF * 2);
    float*  ao  = (float*)alloc(MD * 4);
    float*  x1f = (float*)alloc(MD * 4);
    float*  h2  = (float*)alloc(MD * 4);

    const float ALPHA = 2.2133638f;  // 24^0.25
    dim3 blk(256);

    auto cvt = [&](const float* s, __bf16* d, size_t n) {
        cvt_bf16_kernel<<<dim3((unsigned)((n / 4 + 255) / 256)), blk, 0, stream>>>(s, d, (int)n);
    };
    cvt(states, statesb, MD);
    cvt(Wq, Wqb, WD); cvt(Wk, Wkb, WD); cvt(Wv, Wvb, WD); cvt(Wo, Wob, WD);
    cvt(W1, W1b, WF); cvt(W2, W2b, WF);

    // q gets alpha=0.125: bakes the reference's (0.125/32) q-scale AND the x32 logit scale
    gemm_bf16_kernel<false, true ><<<dim3(12, 32), blk, 0, stream>>>(statesb, Wqb, bq, qb, 4096, 768, 768, 0.125f);
    gemm_bf16_kernel<false, true ><<<dim3(12, 32), blk, 0, stream>>>(statesb, Wkb, bk, kb, 4096, 768, 768, 1.0f);
    gemm_bf16_kernel<false, true ><<<dim3(12, 32), blk, 0, stream>>>(statesb, Wvb, bv, vb, 4096, 768, 768, 1.0f);
    attn_kernel<<<dim3(16, 48), blk, 0, stream>>>(qb, kb, vb, pbias, gw, gb, ga, ctxb);
    gemm_bf16_kernel<false, false><<<dim3(12, 32), blk, 0, stream>>>(ctxb, Wob, bo, ao, 4096, 768, 768, 1.0f);
    add_ln_kernel<true ><<<4096, blk, 0, stream>>>(states, ALPHA, ao, l1g, l1b, x1f, x1b);
    gemm_bf16_kernel<true , true ><<<dim3(48, 32), blk, 0, stream>>>(x1b, W1b, b1, h1b, 4096, 3072, 768, 1.0f);
    gemm_bf16_kernel<false, false><<<dim3(12, 32), blk, 0, stream>>>(h1b, W2b, b2, h2, 4096, 768, 3072, 1.0f);
    float* outp = (float*)d_out;
    add_ln_kernel<false><<<4096, blk, 0, stream>>>(x1f, ALPHA, h2, l2g, l2b, outp, nullptr);
    // tuple output: (out, positional_bias) concatenated flat
    hipMemcpyAsync(outp + MD, pbias, (size_t)48 * 1024 * 1024 * sizeof(float),
                   hipMemcpyDeviceToDevice, stream);
}
